// cross_mamba_72524817760827
// MI455X (gfx1250) — matrix operand order repaired and checked by reading, not compiled, not run on hardware
//
#include <hip/hip_runtime.h>
#include <hip/hip_bf16.h>

#define D_MODEL 256
#define D_INNER 512
#define D_STATE 16
#define DT_RANK 16
#define NCONV   4
#define LSEQ    4096
#define BATCH   4
#define TTOK    (BATCH*LSEQ)

typedef __attribute__((ext_vector_type(16))) __bf16 v16bf;
typedef __attribute__((ext_vector_type(8)))  __bf16 v8bf;
typedef __attribute__((ext_vector_type(8)))  float  v8f;

union FragBF { v16bf v; v8bf h[2]; };

__device__ __forceinline__ float siluf(float x){ return x / (1.f + __expf(-x)); }
__device__ __forceinline__ float softplusf(float x){ return (x > 20.f) ? x : log1pf(__expf(x)); }

// ---------------- weight transpose + f32->bf16 convert: dst[N][Kpad] = src[K][N] ----------------
__global__ void wconvert_kernel(const float* __restrict__ src, __bf16* __restrict__ dst,
                                int K, int N, int Kpad)
{
    int idx = blockIdx.x*blockDim.x + threadIdx.x;
    if (idx >= N*Kpad) return;
    int n = idx / Kpad, k = idx % Kpad;
    float v = (k < K) ? src[(size_t)k*N + n] : 0.f;
    dst[idx] = (__bf16)v;
}

// ---------------- rmsnorm(q)*w_fw / *w_bw -> bf16, v -> bf16 (one wave per token) ---------------
__global__ void rmsnorm_kernel(const float* __restrict__ q, const float* __restrict__ v,
                               const float* __restrict__ wfw, const float* __restrict__ wbw,
                               __bf16* __restrict__ hq_fw, __bf16* __restrict__ hq_bw,
                               __bf16* __restrict__ v_bf)
{
    int wave = blockIdx.x*(blockDim.x>>5) + (threadIdx.x>>5);
    if (wave >= TTOK) return;
    int lane = threadIdx.x & 31;
    const float* qp = q + (size_t)wave*D_MODEL + lane*8;
    const float* vp = v + (size_t)wave*D_MODEL + lane*8;
    float x[8];
    #pragma unroll
    for (int i=0;i<8;i++) x[i] = qp[i];
    float ss = 0.f;
    #pragma unroll
    for (int i=0;i<8;i++) ss += x[i]*x[i];
    #pragma unroll
    for (int m=16;m>=1;m>>=1) ss += __shfl_xor(ss, m, 32);
    float inv = rsqrtf(ss*(1.f/(float)D_MODEL) + 1e-5f);
    size_t base = (size_t)wave*D_MODEL + lane*8;
    #pragma unroll
    for (int i=0;i<8;i++){
        int c = lane*8 + i;
        float hn = x[i]*inv;
        hq_fw[base+i] = (__bf16)(hn*wfw[c]);
        hq_bw[base+i] = (__bf16)(hn*wbw[c]);
        v_bf[base+i]  = (__bf16)vp[i];
    }
}

// ---------------- generic bf16 WMMA GEMM: Dst[M,N] = A[M,K] @ BT[N,K]^T ----------------
// Branchless: N must be a multiple of 16*NT, M multiple of 16, K multiple of 32.
// MODE bit0: softplus(acc + bias[n]); MODE bit1: store bf16 (else f32)
template<int NT, int MODE>
__global__ void gemm_wmma_kernel(const __bf16* __restrict__ A, const __bf16* __restrict__ BT,
                                 void* __restrict__ Dst, const float* __restrict__ bias,
                                 int M, int N, int K)
{
    int lane = threadIdx.x & 31;
    int wave = blockIdx.x*(blockDim.x>>5) + (threadIdx.x>>5);
    int nGroups = (N >> 4)/NT;
    int mT = wave / nGroups, ng = wave % nGroups;
    if (mT >= (M>>4)) return;                 // wave-uniform: EXEC all-ones at every WMMA
    int m0 = mT << 4;
    int lr = lane & 15, hi = lane >> 4;
    v8f acc[NT] = {};
    const __bf16* arow = A + (size_t)(m0 + lr)*K + hi*8;
    const __bf16* brow[NT];
    #pragma unroll
    for (int j = 0; j < NT; j++)
        brow[j] = BT + (size_t)(((ng*NT + j) << 4) + lr)*K + hi*8;
    for (int k0 = 0; k0 < K; k0 += 32){
        FragBF a, b[NT];
        a.h[0] = *(const v8bf*)(arow + k0);        // lanes 0-15: K 0..7   | 16-31: K 8..15
        a.h[1] = *(const v8bf*)(arow + k0 + 16);   // lanes 0-15: K 16..23 | 16-31: K 24..31
        #pragma unroll
        for (int j = 0; j < NT; j++){
            b[j].h[0] = *(const v8bf*)(brow[j] + k0);      // same K split as A
            b[j].h[1] = *(const v8bf*)(brow[j] + k0 + 16);
        }
        __builtin_prefetch(arow + k0 + 32, 0, 1);  // global_prefetch_b8
        #pragma unroll
        for (int j = 0; j < NT; j++)
            acc[j] = __builtin_amdgcn_wmma_f32_16x16x32_bf16(
                         false, a.v, false, b[j].v, (short)0, acc[j], false, false);
    }
    #pragma unroll
    for (int j = 0; j < NT; j++){
        int col = ((ng*NT + j) << 4) + lr;
        int rb  = hi*8;
        #pragma unroll
        for (int r = 0; r < 8; r++){
            float vv = acc[j][r];
            if (MODE & 1) vv = softplusf(vv + bias[col]);
            size_t idx = (size_t)(m0 + rb + r)*N + col;
            if (MODE & 2) ((__bf16*)Dst)[idx] = (__bf16)vv;
            else          ((float*)Dst)[idx]  = vv;
        }
    }
}

// ---------------- causal depthwise conv (width 4) + SiLU, plus silu(z) ----------------
__global__ void conv_silu_kernel(const __bf16* __restrict__ XQZ, const __bf16* __restrict__ XV,
    const float* __restrict__ cqw, const float* __restrict__ cqb,
    const float* __restrict__ cvw, const float* __restrict__ cvb,
    __bf16* __restrict__ xq_s, __bf16* __restrict__ xv_s, __bf16* __restrict__ sz,
    int dirFlip)
{
    int idx = blockIdx.x*blockDim.x + threadIdx.x;
    if (idx >= TTOK*D_INNER) return;
    int c = idx % D_INNER;
    int t = idx / D_INNER;
    int l = t % LSEQ;
    float aq = cqb[c], av = cvb[c];
    #pragma unroll
    for (int j = 0; j < NCONV; j++){
        int lag = NCONV-1-j;
        int ln  = dirFlip ? (l + lag) : (l - lag);   // previous token in scan order
        if (ln < 0 || ln >= LSEQ) continue;
        size_t tn = (size_t)(t - l) + ln;
        aq += cqw[c*NCONV+j] * (float)XQZ[tn*(2*D_INNER) + c];
        av += cvw[c*NCONV+j] * (float)XV [tn*D_INNER + c];
    }
    xq_s[idx] = (__bf16)siluf(aq);
    xv_s[idx] = (__bf16)siluf(av);
    sz[idx]   = (__bf16)siluf((float)XQZ[(size_t)t*(2*D_INNER) + D_INNER + c]);
}

// ---------------- dt_lo (cols 0..15 of dtB) -> bf16 padded to K=32 ----------------
__global__ void dtlo_kernel(const float* __restrict__ dtB, __bf16* __restrict__ dtlo)
{
    int idx = blockIdx.x*blockDim.x + threadIdx.x;
    if (idx >= TTOK*32) return;
    int k = idx & 31;
    dtlo[idx] = (k < DT_RANK) ? (__bf16)dtB[idx] : (__bf16)0.f;
}

// ---------------- selective scan: thread = channel, grid = (batch, channel-chunk) --------------
__global__ void scan_kernel(const __bf16* __restrict__ xv_s, const float* __restrict__ dt,
    const float* __restrict__ dtB, const float* __restrict__ Cc,
    const float* __restrict__ A_log, const float* __restrict__ Dp,
    const __bf16* __restrict__ sz, __bf16* __restrict__ gy, int dirFlip)
{
    int c = blockIdx.y*blockDim.x + threadIdx.x;   // channel
    int b = blockIdx.x;                            // batch
    float A[D_STATE];
    #pragma unroll
    for (int n = 0; n < D_STATE; n++) A[n] = -__expf(A_log[c*D_STATE + n]);
    float Dv = Dp[c];
    float h[D_STATE] = {};
    for (int p = 0; p < LSEQ; p++){
        int l = dirFlip ? (LSEQ-1-p) : p;
        size_t t = (size_t)b*LSEQ + l;
        float dtv = dt[t*D_INNER + c];
        float xv  = (float)xv_s[t*D_INNER + c];
        const float* Bp = dtB + t*32 + DT_RANK;
        const float* Cp = Cc  + t*D_STATE;
        float dx = dtv*xv;
        float y = 0.f;
        #pragma unroll
        for (int n = 0; n < D_STATE; n++){
            float dA = __expf(dtv*A[n]);
            h[n] = dA*h[n] + dx*Bp[n];
            y = fmaf(h[n], Cp[n], y);
        }
        y = fmaf(xv, Dv, y);
        gy[t*D_INNER + c] = (__bf16)(y * (float)sz[t*D_INNER + c]);
    }
}

// ---------------- fused dual out-projection + residual: out = gyf@Wf + gyb@Wb + 2*q -------------
__global__ void outproj_kernel(const __bf16* __restrict__ gyf, const __bf16* __restrict__ gyb,
    const __bf16* __restrict__ WTf, const __bf16* __restrict__ WTb,
    const float* __restrict__ qin, float* __restrict__ out)
{
    const int K = D_INNER, N = D_MODEL, NT = 4;
    int lane = threadIdx.x & 31;
    int wave = blockIdx.x*(blockDim.x>>5) + (threadIdx.x>>5);
    int nGroups = (N >> 4)/NT;                  // 4
    int mT = wave / nGroups, ng = wave % nGroups;
    if (mT >= (TTOK >> 4)) return;
    int m0 = mT << 4;
    int lr = lane & 15, hi = lane >> 4;
    v8f acc[NT] = {};
    const __bf16* af = gyf + (size_t)(m0 + lr)*K + hi*8;
    const __bf16* ab = gyb + (size_t)(m0 + lr)*K + hi*8;
    const __bf16* bf[NT]; const __bf16* bb[NT];
    #pragma unroll
    for (int j = 0; j < NT; j++){
        size_t nb = (size_t)(((ng*NT + j) << 4) + lr)*K + hi*8;
        bf[j] = WTf + nb;
        bb[j] = WTb + nb;
    }
    for (int k0 = 0; k0 < K; k0 += 32){
        FragBF a1, a2;
        a1.h[0] = *(const v8bf*)(af + k0);  a1.h[1] = *(const v8bf*)(af + k0 + 16);
        a2.h[0] = *(const v8bf*)(ab + k0);  a2.h[1] = *(const v8bf*)(ab + k0 + 16);
        __builtin_prefetch(af + k0 + 32, 0, 1);
        #pragma unroll
        for (int j = 0; j < NT; j++){
            FragBF b1, b2;
            b1.h[0] = *(const v8bf*)(bf[j] + k0);  b1.h[1] = *(const v8bf*)(bf[j] + k0 + 16);
            b2.h[0] = *(const v8bf*)(bb[j] + k0);  b2.h[1] = *(const v8bf*)(bb[j] + k0 + 16);
            acc[j] = __builtin_amdgcn_wmma_f32_16x16x32_bf16(false, a1.v, false, b1.v, (short)0, acc[j], false, false);
            acc[j] = __builtin_amdgcn_wmma_f32_16x16x32_bf16(false, a2.v, false, b2.v, (short)0, acc[j], false, false);
        }
    }
    #pragma unroll
    for (int j = 0; j < NT; j++){
        int col = ((ng*NT + j) << 4) + lr;
        int rb  = hi*8;
        #pragma unroll
        for (int r = 0; r < 8; r++){
            size_t idx = (size_t)(m0 + rb + r)*N + col;
            out[idx] = acc[j][r] + 2.f*qin[idx];
        }
    }
}

extern "C" void kernel_launch(void* const* d_in, const int* in_sizes, int n_in,
                              void* d_out, int out_size, void* d_ws, size_t ws_size,
                              hipStream_t stream)
{
    (void)in_sizes; (void)n_in; (void)out_size; (void)ws_size;
    const float* q = (const float*)d_in[0];
    const float* v = (const float*)d_in[1];
    // param leaves in _make_params insertion order, fw then bw:
    // 0 norm_w 1 W_in_q 2 W_in_v 3 conv_q_w 4 conv_q_b 5 conv_v_w 6 conv_v_b
    // 7 W_xproj_v 8 W_xproj_q 9 W_dt 10 b_dt 11 A_log 12 D 13 W_out
    auto P = [&](int dir, int j)->const float* { return (const float*)d_in[2 + dir*14 + j]; };

    char* wsp = (char*)d_ws;
    size_t off = 0;
    auto alloc = [&](size_t bytes)->void* {
        void* p = wsp + off;
        off += (bytes + 255) & ~(size_t)255;
        return p;
    };
    const size_t T = TTOK;
    __bf16 *WinqT[2], *WinvT[2], *WxvT[2], *WxqT[2], *WdtT[2], *WoutT[2];
    __bf16 *hq[2], *XQZ[2], *XVb[2], *xq_s[2], *xv_s[2], *szb[2], *dtlo[2], *gy[2], *v_bf;
    float  *dtB[2], *Cc[2], *dtf[2];
    for (int d = 0; d < 2; d++){
        WinqT[d] = (__bf16*)alloc(1024*256*2);
        WinvT[d] = (__bf16*)alloc(512*256*2);
        WxvT[d]  = (__bf16*)alloc(32*512*2);
        WxqT[d]  = (__bf16*)alloc(16*512*2);
        WdtT[d]  = (__bf16*)alloc(512*32*2);
        WoutT[d] = (__bf16*)alloc(256*512*2);
        hq[d]    = (__bf16*)alloc(T*256*2);
        XQZ[d]   = (__bf16*)alloc(T*1024*2);
        XVb[d]   = (__bf16*)alloc(T*512*2);
        xq_s[d]  = (__bf16*)alloc(T*512*2);
        xv_s[d]  = (__bf16*)alloc(T*512*2);
        szb[d]   = (__bf16*)alloc(T*512*2);
        dtlo[d]  = (__bf16*)alloc(T*32*2);
        gy[d]    = (__bf16*)alloc(T*512*2);
        dtB[d]   = (float*)alloc(T*32*4);
        Cc[d]    = (float*)alloc(T*16*4);
        dtf[d]   = (float*)alloc(T*512*4);
    }
    v_bf = (__bf16*)alloc(T*256*2);

    dim3 blk(256);
    auto grid1 = [](size_t n, size_t per){ return dim3((unsigned)((n + per - 1)/per)); };
    auto gemmGrid = [](int M, int N, int NT){
        int nGroups = (N >> 4)/NT;
        size_t waves = (size_t)(M >> 4)*nGroups;
        return dim3((unsigned)((waves + 7)/8));
    };

    // 0) weight transpose/convert (tiny)
    for (int d = 0; d < 2; d++){
        wconvert_kernel<<<grid1(1024*256,256), blk, 0, stream>>>(P(d,1),  WinqT[d], 256, 1024, 256);
        wconvert_kernel<<<grid1(512*256,256),  blk, 0, stream>>>(P(d,2),  WinvT[d], 256, 512, 256);
        wconvert_kernel<<<grid1(32*512,256),   blk, 0, stream>>>(P(d,7),  WxvT[d],  512, 32,  512);
        wconvert_kernel<<<grid1(16*512,256),   blk, 0, stream>>>(P(d,8),  WxqT[d],  512, 16,  512);
        wconvert_kernel<<<grid1(512*32,256),   blk, 0, stream>>>(P(d,9),  WdtT[d],  16,  512, 32);
        wconvert_kernel<<<grid1(256*512,256),  blk, 0, stream>>>(P(d,13), WoutT[d], 512, 256, 512);
    }
    // 1) rmsnorm + bf16 convert (one wave/token, 8 waves/block)
    rmsnorm_kernel<<<grid1(T,8), blk, 0, stream>>>(q, v, P(0,0), P(1,0), hq[0], hq[1], v_bf);

    for (int d = 0; d < 2; d++){
        // 2) input projections (WMMA, bf16 out)
        gemm_wmma_kernel<4,2><<<gemmGrid((int)T,1024,4), blk, 0, stream>>>(hq[d], WinqT[d], XQZ[d], nullptr, (int)T, 1024, 256);
        gemm_wmma_kernel<4,2><<<gemmGrid((int)T,512,4),  blk, 0, stream>>>(v_bf,  WinvT[d], XVb[d], nullptr, (int)T, 512,  256);
        // 3) causal dwconv + silu (+ silu(z))
        conv_silu_kernel<<<grid1(T*512,256), blk, 0, stream>>>(XQZ[d], XVb[d], P(d,3), P(d,4), P(d,5), P(d,6),
                                                               xq_s[d], xv_s[d], szb[d], d);
        // 4) x-projections (WMMA, f32 out)
        gemm_wmma_kernel<2,0><<<gemmGrid((int)T,32,2), blk, 0, stream>>>(xv_s[d], WxvT[d], dtB[d], nullptr, (int)T, 32, 512);
        gemm_wmma_kernel<1,0><<<gemmGrid((int)T,16,1), blk, 0, stream>>>(xq_s[d], WxqT[d], Cc[d],  nullptr, (int)T, 16, 512);
        dtlo_kernel<<<grid1(T*32,256), blk, 0, stream>>>(dtB[d], dtlo[d]);
        // 5) dt = softplus(dt_lo@W_dt + b_dt) (WMMA, K padded to 32, fused epilogue)
        gemm_wmma_kernel<4,1><<<gemmGrid((int)T,512,4), blk, 0, stream>>>(dtlo[d], WdtT[d], dtf[d], P(d,10), (int)T, 512, 32);
        // 6) selective scan (serial in L; 4 batches x 4 channel-chunks x 128 threads)
        scan_kernel<<<dim3(BATCH,4), dim3(128), 0, stream>>>(xv_s[d], dtf[d], dtB[d], Cc[d],
                                                             P(d,11), P(d,12), szb[d], gy[d], d);
    }
    // 7) fused out-projection for both directions + 2*q residual
    outproj_kernel<<<gemmGrid((int)T,256,4), blk, 0, stream>>>(gy[0], gy[1], WoutT[0], WoutT[1], q, (float*)d_out);
}
